// AnalyzableUniformAttention_14448269983911
// MI455X (gfx1250) — matrix operand order repaired and checked by reading, not compiled, not run on hardware
//
#include <hip/hip_runtime.h>
#include <hip/hip_bf16.h>
#include <stdint.h>

typedef __attribute__((ext_vector_type(16))) __bf16       v16bf;
typedef __attribute__((ext_vector_type(8)))  float        v8f;
typedef __attribute__((ext_vector_type(8)))  unsigned int u32x8;
typedef __attribute__((ext_vector_type(4)))  unsigned int u32x4;
typedef __attribute__((ext_vector_type(4)))  float        f32x4;

#define HDIM   256
#define LSEQ   4096
#define BATCH  16
#define MROWS  (BATCH * LSEQ)     // 65536 rows total
#define CHUNK  128                // rows per scan chunk (== block M-tile)
#define NCHUNK (MROWS / CHUNK)    // 512

// ---- bf16 helpers (round-to-nearest-even) ----
__device__ __forceinline__ uint32_t bf16_rne(float f) {
  uint32_t u = __builtin_bit_cast(uint32_t, f);
  return (u + 0x7FFFu + ((u >> 16) & 1u)) >> 16;
}
__device__ __forceinline__ float bf16_to_f32(uint32_t h) {
  return __builtin_bit_cast(float, h << 16);
}

// ============================================================================
// Kernel 1: Weff[k][n] = sum_a Wv[a][k] * Wo[n][a]  (f32), then split into
// hi/lo bf16 planes, stored pre-swizzled in WMMA B-fragment order:
//   ws layout: [nh(2)][kc(8)][ntl(8)][plane(2)][lane(32)][16 bf16]
//   lane = khalf*16 + nn holds K = kc*32 + khalf*16 + j (j=0..15), N = base+nn
// ============================================================================
__global__ void __launch_bounds__(256) weff_kernel(const float* __restrict__ Wv,
                                                   const float* __restrict__ Wo,
                                                   uint16_t* __restrict__ wbuf) {
  __shared__ float tWv[256 * 16];   // Wv[t][k0..k0+16)
  __shared__ float tWo[16 * 256];   // Wo[n0..n0+16)[t]
  const int tid = threadIdx.x;
  const int bx  = blockIdx.x;
  const int khalf = bx & 1;
  const int ntl   = (bx >> 1) & 7;
  const int kc    = (bx >> 4) & 7;
  const int nh    = bx >> 7;
  const int k0 = kc * 32 + khalf * 16;
  const int n0 = nh * 128 + ntl * 16;

  { // each thread loads 16 floats of one Wv row (the k-slice)
    const f32x4* s = (const f32x4*)(Wv + (size_t)tid * HDIM + k0);
    f32x4* d = (f32x4*)(tWv + tid * 16);
    d[0] = s[0]; d[1] = s[1]; d[2] = s[2]; d[3] = s[3];
  }
  { // 16 rows x 256 cols of Wo, 16 contiguous floats per thread
    const int row = tid >> 4, seg = tid & 15;
    const f32x4* s = (const f32x4*)(Wo + (size_t)(n0 + row) * HDIM + seg * 16);
    f32x4* d = (f32x4*)(tWo + row * 256 + seg * 16);
    d[0] = s[0]; d[1] = s[1]; d[2] = s[2]; d[3] = s[3];
  }
  __syncthreads();

  const int kl = tid & 15, nl = tid >> 4;
  float acc = 0.f;
#pragma unroll 8
  for (int t = 0; t < 256; ++t)
    acc = fmaf(tWv[t * 16 + kl], tWo[nl * 256 + t], acc);

  const uint32_t hi = bf16_rne(acc);
  const uint32_t lo = bf16_rne(acc - bf16_to_f32(hi));
  const size_t base_e = (size_t)nh * 65536 + (size_t)(kc * 8 + ntl) * 1024;
  const int lane_s = khalf * 16 + nl;
  wbuf[base_e + lane_s * 16 + kl]       = (uint16_t)hi;   // hi plane
  wbuf[base_e + 512 + lane_s * 16 + kl] = (uint16_t)lo;   // lo plane
}

// ============================================================================
// Kernel 2: v = x @ Weff via v_wmma_f32_16x16x32_bf16 (bf16x3 split product).
// Block = 256 thr (8 waves). Each block owns one 128-column half of Weff
// (both planes, 128KB) in LDS (filled via async global->LDS DMA path) and
// iterates 4 M-tiles of 128 rows. Emits deterministic per-chunk column sums.
// ============================================================================
__global__ void __launch_bounds__(256)
__attribute__((amdgpu_waves_per_eu(4)))
gemm_kernel(const float* __restrict__ x,
            const uint16_t* __restrict__ wbuf,
            float* __restrict__ vout,
            float* __restrict__ csum) {
  extern __shared__ char smem[];                 // 131072 (Weff) + 4096 (wred)
  float* wred = (float*)(smem + 131072);
  const int tid  = threadIdx.x;
  const int lane = tid & 31, wave = tid >> 5;
  const int nh   = blockIdx.x & 1;
  const int mgrp = blockIdx.x >> 1;

  { // 128KB Weff-half fill via async global->LDS (ASYNCcnt path, no VGPR data)
    const char* gsrc = (const char*)wbuf + (size_t)nh * 131072;
    const uint32_t lbase = (uint32_t)(uintptr_t)&smem[0];
#pragma unroll 4
    for (int i = 0; i < 32; ++i) {
      const uint32_t off = (uint32_t)(tid + i * 256) * 16u;
      const uint64_t g = (uint64_t)(uintptr_t)(gsrc + off);
      const uint32_t l = lbase + off;
      asm volatile("global_load_async_to_lds_b128 %0, %1, off"
                   :: "v"(l), "v"(g) : "memory");
    }
    asm volatile("s_wait_asynccnt 0x0" ::: "memory");
  }
  __syncthreads();

  const int g    = lane >> 4;      // lane half (A-layout K grouping)
  const int mloc = lane & 15;      // A-layout: M = lane % 16

  for (int it = 0; it < 4; ++it) {
    const int mblk = mgrp + it * 128;           // chunk id, 0..511
    const int rowW = mblk * CHUNK + wave * 16;  // this wave's 16-row tile
    v8f acc[8] = {};

    // kc NOT fully unrolled: keeps live B-fragments bounded so the allocator
    // stays <=256 VGPRs (avoids s_set_vgpr_msb overhead, allows 4 waves/SIMD)
#pragma unroll 1
    for (int kc = 0; kc < 8; ++kc) {
      // ---- A fragment: row rowW+mloc, K chunk kc*32, hi/lo bf16 split ----
      // lane half g holds K in {16g..16g+15} (contiguous, matches B packing)
      const float* xr = x + (size_t)(rowW + mloc) * HDIM + kc * 32 + g * 16;
      const f32x4 c0 = *(const f32x4*)(xr);
      const f32x4 c1 = *(const f32x4*)(xr + 4);
      const f32x4 c2 = *(const f32x4*)(xr + 8);
      const f32x4 c3 = *(const f32x4*)(xr + 12);
      const float f[16] = {c0.x, c0.y, c0.z, c0.w, c1.x, c1.y, c1.z, c1.w,
                           c2.x, c2.y, c2.z, c2.w, c3.x, c3.y, c3.z, c3.w};
      u32x8 uh, ul;
#pragma unroll
      for (int p = 0; p < 8; ++p) {
        const uint32_t h0 = bf16_rne(f[2 * p]);
        const uint32_t h1 = bf16_rne(f[2 * p + 1]);
        const uint32_t l0 = bf16_rne(f[2 * p]     - bf16_to_f32(h0));
        const uint32_t l1 = bf16_rne(f[2 * p + 1] - bf16_to_f32(h1));
        uh[p] = h0 | (h1 << 16);
        ul[p] = l0 | (l1 << 16);
      }
      const v16bf Ah = __builtin_bit_cast(v16bf, uh);
      const v16bf Al = __builtin_bit_cast(v16bf, ul);

#pragma unroll
      for (int ntl = 0; ntl < 8; ++ntl) {
        const char* fb = smem + ((kc * 8 + ntl) << 11) + lane * 32;
        const v16bf Bh = __builtin_bit_cast(v16bf, *(const u32x8*)fb);
        const v16bf Bl = __builtin_bit_cast(v16bf, *(const u32x8*)(fb + 1024));
        // D = xh*Wh + xh*Wl + xl*Wh + C  (bf16x3 ~ f32 accuracy)
        acc[ntl] = __builtin_amdgcn_wmma_f32_16x16x32_bf16(
            false, Ah, false, Bh, (short)0, acc[ntl], false, false);
        acc[ntl] = __builtin_amdgcn_wmma_f32_16x16x32_bf16(
            false, Ah, false, Bl, (short)0, acc[ntl], false, false);
        acc[ntl] = __builtin_amdgcn_wmma_f32_16x16x32_bf16(
            false, Al, false, Bh, (short)0, acc[ntl], false, false);
      }
    }

    // ---- store v (C/D layout: VGPR r -> row r | 8+r, col = lane&15) ----
    const int nn = lane & 15;
    const int m0 = rowW + g * 8;
    const size_t colBase = (size_t)nh * 128;
#pragma unroll
    for (int ntl = 0; ntl < 8; ++ntl) {
#pragma unroll
      for (int r = 0; r < 8; ++r)
        vout[(size_t)(m0 + r) * HDIM + colBase + ntl * 16 + nn] = acc[ntl][r];
    }

    // ---- deterministic chunk column sums (no FP atomics) ----
#pragma unroll
    for (int ntl = 0; ntl < 8; ++ntl) {
      float s = acc[ntl][0] + acc[ntl][1] + acc[ntl][2] + acc[ntl][3] +
                acc[ntl][4] + acc[ntl][5] + acc[ntl][6] + acc[ntl][7];
      s += __shfl_xor(s, 16, 32);              // add complementary row half
      if (lane < 16) wred[wave * 128 + ntl * 16 + lane] = s;
    }
    __syncthreads();
    if (tid < 128) {
      float t = 0.f;
#pragma unroll
      for (int w = 0; w < 8; ++w) t += wred[w * 128 + tid];
      csum[(size_t)mblk * HDIM + nh * 128 + tid] = t;
    }
    __syncthreads();
  }
}

// ============================================================================
// Kernel 3: exclusive scan of chunk sums along L (32 chunks per batch image)
// ============================================================================
__global__ void __launch_bounds__(256) scan_kernel(const float* __restrict__ csum,
                                                   float* __restrict__ offs) {
  const int h = threadIdx.x;
  const int b = blockIdx.x;
  float run = 0.f;
  for (int c = 0; c < LSEQ / CHUNK; ++c) {
    const size_t idx = (size_t)(b * (LSEQ / CHUNK) + c) * HDIM + h;
    offs[idx] = run;
    run += csum[idx];
  }
}

// ============================================================================
// Kernel 4: in-place causal cumulative mean + residual.
// out[row,h] = x[row,h] + (offset + running-sum of v up to row) / (l+1)
// (v lives in d_out; each element is read before being overwritten)
// ============================================================================
__global__ void __launch_bounds__(256) finalize_kernel(const float* __restrict__ x,
                                                       const float* __restrict__ offs,
                                                       float* __restrict__ out) {
  const int h = threadIdx.x;
  const int chunk = blockIdx.x;
  float run = offs[(size_t)chunk * HDIM + h];
  const size_t base = (size_t)chunk * CHUNK * HDIM + h;
  const int l0 = (chunk * CHUNK) & (LSEQ - 1);
  const float* xp = x + base;
  float* op = out + base;
  for (int i = 0; i < CHUNK; ++i) {
    __builtin_prefetch(xp + (size_t)(i + 16) * HDIM, 0, 0);
    const float v = op[(size_t)i * HDIM];
    run += v;
    const float inv = 1.0f / (float)(l0 + i + 1);
    op[(size_t)i * HDIM] = xp[(size_t)i * HDIM] + run * inv;
  }
}

// ============================================================================
extern "C" void kernel_launch(void* const* d_in, const int* in_sizes, int n_in,
                              void* d_out, int out_size, void* d_ws, size_t ws_size,
                              hipStream_t stream) {
  const float* x  = (const float*)d_in[0];
  const float* Wv = (const float*)d_in[1];
  const float* Wo = (const float*)d_in[2];
  float* out = (float*)d_out;

  // workspace: [0,256KB) swizzled Weff bf16 hi/lo; then chunk sums; then offsets
  uint16_t* wbuf = (uint16_t*)d_ws;
  float* csum = (float*)((char*)d_ws + 262144);
  float* offs = (float*)((char*)d_ws + 262144 + 524288);

  (void)hipFuncSetAttribute((const void*)gemm_kernel,
                            hipFuncAttributeMaxDynamicSharedMemorySize, 135168);

  weff_kernel<<<256, 256, 0, stream>>>(Wv, Wo, wbuf);
  gemm_kernel<<<256, 256, 135168, stream>>>(x, wbuf, out, csum);
  scan_kernel<<<BATCH, 256, 0, stream>>>(csum, offs);
  finalize_kernel<<<NCHUNK, 256, 0, stream>>>(x, offs, out);
}
